// ArcFaceLoss_28183575396662
// MI455X (gfx1250) — matrix operand order repaired and checked by reading, not compiled, not run on hardware
//
#include <hip/hip_runtime.h>

typedef __attribute__((ext_vector_type(16))) _Float16 v16h;
typedef __attribute__((ext_vector_type(8)))  _Float16 v8h;
typedef __attribute__((ext_vector_type(8)))  float    v8f;

#define B_ROWS 1024
#define DIM    512
#define NCLS   50000
#define CHUNK  512
#define NCHUNKS 98                 // 98*512 = 50176 >= 50000
#define CPAD   (NCHUNKS * CHUNK)
#define TILES  (CHUNK / 16)        // 32 N-tiles per chunk
#define KSTEPS (DIM / 32)          // 16 WMMA k-steps over K=512
#define MWAVES 8                   // waves per block
#define MROWS  256                 // rows per block: 8 waves x 2 M-tiles x 16
#define TILE_BYTES (16 * DIM * 2)  // one 16-col B tile in LDS = 16 KB
#define SCALE_F 64.0f

#define COS_M_F 0.8775825618903728f    // cos(0.5)
#define SIN_M_F 0.4794255386042030f    // sin(0.5)
#define TH_F   (-0.8775825618903728f)  // cos(pi - 0.5)
#define MM_F    0.2397127693021015f    // sin(pi - 0.5) * 0.5

#define NEG_INF (-__builtin_inff())

// -------- 1/2: row L2-normalize (fp32 -> f16), zero-pad rows >= validRows ----
__global__ __launch_bounds__(256)
void normalize_rows(const float* __restrict__ src, _Float16* __restrict__ dst,
                    int nRows, int validRows) {
  const int lane = threadIdx.x & 31;
  const int row  = blockIdx.x * 8 + (threadIdx.x >> 5);
  if (row >= nRows) return;
  _Float16* drow = dst + (size_t)row * DIM;
  if (row >= validRows) {
    for (int d = lane; d < DIM; d += 32) drow[d] = (_Float16)0.0f;
    return;
  }
  const float* srow = src + (size_t)row * DIM;
  float ss = 0.f;
  for (int d = lane; d < DIM; d += 32) { float v = srow[d]; ss += v * v; }
  #pragma unroll
  for (int off = 16; off >= 1; off >>= 1) ss += __shfl_xor(ss, off, 32);
  float inv = 1.0f / fmaxf(sqrtf(ss), 1e-12f);
  for (int d = lane; d < DIM; d += 32) drow[d] = (_Float16)(srow[d] * inv);
}

// -------- 3: per-row label logits: SCALE*phi and SCALE*cos (1 wave / row) ---
__global__ __launch_bounds__(256)
void label_phi(const _Float16* __restrict__ eN, const _Float16* __restrict__ wN,
               const int* __restrict__ labels,
               float* __restrict__ phil, float* __restrict__ cosl) {
  const int lane = threadIdx.x & 31;
  const int b    = blockIdx.x * 8 + (threadIdx.x >> 5);
  if (b >= B_ROWS) return;
  const int lbl = labels[b];
  const _Float16* er = eN + (size_t)b   * DIM;
  const _Float16* wr = wN + (size_t)lbl * DIM;
  float dot = 0.f;
  for (int d = lane; d < DIM; d += 32) dot += (float)er[d] * (float)wr[d];
  #pragma unroll
  for (int off = 16; off >= 1; off >>= 1) dot += __shfl_xor(dot, off, 32);
  if (lane == 0) {
    float c  = dot;
    float c2 = fminf(fmaxf(c * c, 0.f), 1.f);
    float s  = sqrtf(1.f - c2);
    float phi = c * COS_M_F - s * SIN_M_F;
    phi = (c > TH_F) ? phi : (c - MM_F);
    phil[b] = SCALE_F * phi;
    cosl[b] = SCALE_F * c;
  }
}

// Issue one 16-col x 512-K B tile (16 KB) as 4 async b128 loads per thread
// slice; the 24-bit instruction offset shifts BOTH the global and LDS address.
__device__ __forceinline__ void issue_tile_async(const _Float16* sbase,
                                                 unsigned vOffBytes,
                                                 unsigned ldsDst) {
  asm volatile("global_load_async_to_lds_b128 %0, %1, %2 offset:0"
               :: "v"(ldsDst), "v"(vOffBytes), "s"(sbase) : "memory");
  asm volatile("global_load_async_to_lds_b128 %0, %1, %2 offset:16"
               :: "v"(ldsDst), "v"(vOffBytes), "s"(sbase) : "memory");
  asm volatile("global_load_async_to_lds_b128 %0, %1, %2 offset:32"
               :: "v"(ldsDst), "v"(vOffBytes), "s"(sbase) : "memory");
  asm volatile("global_load_async_to_lds_b128 %0, %1, %2 offset:48"
               :: "v"(ldsDst), "v"(vOffBytes), "s"(sbase) : "memory");
}

// Load one 16x32 A fragment (contiguous k = 16*hi + s, matching B) for row block m0.
__device__ __forceinline__ v16h load_afrag(const _Float16* eN, int m0, int l15,
                                           int hi, int ks) {
  const _Float16* ep = eN + (size_t)(m0 + l15) * DIM + ks * 32;
  v8h lo = *(const v8h*)(ep + 16 * hi);
  v8h hh = *(const v8h*)(ep + 16 * hi + 8);
  v16h a;
  ((v8h*)&a)[0] = lo;
  ((v8h*)&a)[1] = hh;
  return a;
}

// -------- 4: fused f16 WMMA GEMM + online logsumexp partials ----------------
// 8 waves / block; each wave owns TWO 16-row M-tiles (block M = 256) so every
// B fragment read from LDS feeds two WMMAs (halves LDS read bandwidth, the
// dominant cost). B tiles staged global->LDS with async DMA, double buffered.
__global__ __launch_bounds__(256)
void arcface_gemm_partial(const _Float16* __restrict__ eN,
                          const _Float16* __restrict__ wN,
                          float* __restrict__ partials) {
  __shared__ __align__(64) _Float16 ldsB[2][16 * DIM];   // 2 x 16 KB

  const int tid  = threadIdx.x;
  const int lane = tid & 31;
  const int wave = tid >> 5;                 // 0..7
  const int l15  = lane & 15;
  const int hi   = lane >> 4;
  const int m0a  = blockIdx.x * MROWS + wave * 16;         // M-tile 0
  const int m0b  = m0a + 128;                              // M-tile 1
  const int chunkStart = blockIdx.y * CHUNK;

  // Copy-slice mapping: thread copies 64 B of column (tid>>4), segment (tid&15).
  const unsigned sliceOff =
      (unsigned)((((tid >> 4) * DIM) + (tid & 15) * 32) * 2);
  const unsigned ldsBase = (unsigned)(unsigned long long)(&ldsB[0][0]);

  // Preload A fragments for both M-tiles: 2 x 16 x v16h = 256 VGPRs/lane.
  v16h afragA[KSTEPS], afragB[KSTEPS];
  #pragma unroll
  for (int ks = 0; ks < KSTEPS; ++ks) afragA[ks] = load_afrag(eN, m0a, l15, hi, ks);
  #pragma unroll
  for (int ks = 0; ks < KSTEPS; ++ks) afragB[ks] = load_afrag(eN, m0b, l15, hi, ks);

  float runMaxA[8], runSumA[8], runMaxB[8], runSumB[8];
  #pragma unroll
  for (int r = 0; r < 8; ++r) {
    runMaxA[r] = NEG_INF; runSumA[r] = 0.f;
    runMaxB[r] = NEG_INF; runSumB[r] = 0.f;
  }

  // Prologue: tile 0 -> buffer 0.
  issue_tile_async(wN + (size_t)chunkStart * DIM, sliceOff, ldsBase + sliceOff);

  for (int t = 0; t < TILES; ++t) {
    const int buf = t & 1;
    if (t + 1 < TILES) {
      // Keep the next tile in flight while we compute this one.
      issue_tile_async(wN + (size_t)(chunkStart + (t + 1) * 16) * DIM, sliceOff,
                       ldsBase + (unsigned)((t + 1) & 1) * TILE_BYTES + sliceOff);
      asm volatile("s_wait_asynccnt 4" ::: "memory");   // tile t landed
    } else {
      asm volatile("s_wait_asynccnt 0" ::: "memory");
    }
    __syncthreads();   // all waves' DMA slices of tile t visible

    const _Float16* bp = &ldsB[buf][l15 * DIM + 16 * hi];
    v8f accA = {}, accB = {};
    #pragma unroll
    for (int ks = 0; ks < KSTEPS; ++ks) {
      v16h b = *(const v16h*)(bp + ks * 32);   // one read feeds two WMMAs
      accA = __builtin_amdgcn_wmma_f32_16x16x32_f16(
          false, afragA[ks], false, b, (short)0, accA, false, false);
      accB = __builtin_amdgcn_wmma_f32_16x16x32_f16(
          false, afragB[ks], false, b, (short)0, accB, false, false);
    }

    // acc[r] = cosine[m0 + r + 8*hi][n0 + l15]; online logsumexp update.
    const int col = chunkStart + t * 16 + l15;
    if (col < NCLS) {
      #pragma unroll
      for (int r = 0; r < 8; ++r) {
        float x  = SCALE_F * accA[r];
        float mo = runMaxA[r];
        float mn = fmaxf(mo, x);
        runSumA[r] = runSumA[r] * __expf(mo - mn) + __expf(x - mn);
        runMaxA[r] = mn;
        float x2  = SCALE_F * accB[r];
        float mo2 = runMaxB[r];
        float mn2 = fmaxf(mo2, x2);
        runSumB[r] = runSumB[r] * __expf(mo2 - mn2) + __expf(x2 - mn2);
        runMaxB[r] = mn2;
      }
    }
    __syncthreads();   // everyone done reading buf before tile t+2 overwrites
  }

  // Merge across the 16 lanes of each half-wave (rows are shared per half).
  #pragma unroll
  for (int off = 8; off >= 1; off >>= 1) {
    #pragma unroll
    for (int r = 0; r < 8; ++r) {
      float mo = __shfl_xor(runMaxA[r], off, 32);
      float so = __shfl_xor(runSumA[r], off, 32);
      float mn = fmaxf(runMaxA[r], mo);
      float s  = 0.f;
      if (mn > NEG_INF)
        s = runSumA[r] * __expf(runMaxA[r] - mn) + so * __expf(mo - mn);
      runMaxA[r] = mn; runSumA[r] = s;

      float mo2 = __shfl_xor(runMaxB[r], off, 32);
      float so2 = __shfl_xor(runSumB[r], off, 32);
      float mn2 = fmaxf(runMaxB[r], mo2);
      float s2  = 0.f;
      if (mn2 > NEG_INF)
        s2 = runSumB[r] * __expf(runMaxB[r] - mn2) + so2 * __expf(mo2 - mn2);
      runMaxB[r] = mn2; runSumB[r] = s2;
    }
  }

  if (l15 == 0) {
    #pragma unroll
    for (int r = 0; r < 8; ++r) {
      const int rowA = m0a + hi * 8 + r;
      const size_t idxA = ((size_t)rowA * NCHUNKS + blockIdx.y) * 2;
      partials[idxA]     = runMaxA[r];
      partials[idxA + 1] = runSumA[r];
      const int rowB = m0b + hi * 8 + r;
      const size_t idxB = ((size_t)rowB * NCHUNKS + blockIdx.y) * 2;
      partials[idxB]     = runMaxB[r];
      partials[idxB + 1] = runSumB[r];
    }
  }
}

// -------- 5: merge chunk partials, swap label logit, mean NLL ---------------
__global__ __launch_bounds__(256)
void finalize(const float* __restrict__ partials,
              const float* __restrict__ phil, const float* __restrict__ cosl,
              float* __restrict__ out) {
  __shared__ float red[256];
  float local = 0.f;
  for (int b = threadIdx.x; b < B_ROWS; b += 256) {
    const float* p = partials + (size_t)b * NCHUNKS * 2;
    float M = NEG_INF;
    for (int c = 0; c < NCHUNKS; ++c) M = fmaxf(M, p[2 * c]);
    float S = 0.f;
    for (int c = 0; c < NCHUNKS; ++c) S += p[2 * c + 1] * __expf(p[2 * c] - M);
    // Replace the label column's cosine logit with the phi logit.
    // phi <= cosine <= M/SCALE, so M is unchanged.
    S += __expf(phil[b] - M) - __expf(cosl[b] - M);
    local += (M + __logf(S)) - phil[b];
  }
  red[threadIdx.x] = local;
  __syncthreads();
  for (int s = 128; s > 0; s >>= 1) {
    if (threadIdx.x < s) red[threadIdx.x] += red[threadIdx.x + s];
    __syncthreads();
  }
  if (threadIdx.x == 0) out[0] = red[0] / (float)B_ROWS;
}

extern "C" void kernel_launch(void* const* d_in, const int* in_sizes, int n_in,
                              void* d_out, int out_size, void* d_ws, size_t ws_size,
                              hipStream_t stream) {
  const float* emb    = (const float*)d_in[0];
  const int*   labels = (const int*)d_in[1];
  const float* weight = (const float*)d_in[2];

  char* ws = (char*)d_ws;
  _Float16* wN = (_Float16*)ws;                                   // CPAD*DIM f16
  _Float16* eN = (_Float16*)(ws + (size_t)CPAD * DIM * 2);        // B*DIM f16
  float* partials = (float*)(ws + (size_t)CPAD * DIM * 2
                                + (size_t)B_ROWS * DIM * 2);      // B*NCHUNKS*2
  float* phil = partials + (size_t)B_ROWS * NCHUNKS * 2;          // B
  float* cosl = phil + B_ROWS;                                    // B

  normalize_rows<<<CPAD / 8, 256, 0, stream>>>(weight, wN, CPAD, NCLS);
  normalize_rows<<<B_ROWS / 8, 256, 0, stream>>>(emb, eN, B_ROWS, B_ROWS);
  label_phi<<<B_ROWS / 8, 256, 0, stream>>>(eN, wN, labels, phil, cosl);
  arcface_gemm_partial<<<dim3(B_ROWS / MROWS, NCHUNKS), 256, 0, stream>>>(
      eN, wN, partials);
  finalize<<<1, 256, 0, stream>>>(partials, phil, cosl, (float*)d_out);
}
